// MultiHeadAttentionWithRPR_21449066676705
// MI455X (gfx1250) — matrix operand order repaired and checked by reading, not compiled, not run on hardware
//
#include <hip/hip_runtime.h>
#include <hip/hip_bf16.h>

typedef __bf16 bf16_t;
typedef __attribute__((ext_vector_type(16))) __bf16 v16bf;
typedef __attribute__((ext_vector_type(8)))  __bf16 v8bf;
typedef __attribute__((ext_vector_type(4)))  __bf16 v4bf;
typedef __attribute__((ext_vector_type(8)))  float  v8f;
typedef __attribute__((ext_vector_type(4)))  float  v4f;

#define BB 4
#define SS 1024
#define DM 1024
#define HH 16
#define HD 64
#define NEGI -1.0e6f

__device__ __forceinline__ v16bf cat8(v8bf lo, v8bf hi) {
  return __builtin_shufflevector(lo, hi, 0,1,2,3,4,5,6,7,8,9,10,11,12,13,14,15);
}
__device__ __forceinline__ v8bf ldg8(const bf16_t* p) { return *(const v8bf*)p; }

__device__ __forceinline__ v8f wmma_bf16(v16bf a, v16bf b, v8f c) {
  return __builtin_amdgcn_wmma_f32_16x16x32_bf16(false, a, false, b, (short)0, c, false, false);
}

// A fragment: row-major row pointer, K-chunk kk. per-lane: 8 bf16 @ kk+half*8, 8 @ +16
__device__ __forceinline__ v16bf lda_frag(const bf16_t* row, int kk, int half) {
  int ab = kk + half * 8;
  return cat8(ldg8(row + ab), ldg8(row + ab + 16));
}
// B fragment: column n's K-major data is contiguous (row n of the [N x K] matrix)
__device__ __forceinline__ v16bf ldb_frag(const bf16_t* row, int kk, int half) {
  int bb = kk + half * 8;
  return cat8(ldg8(row + bb), ldg8(row + bb + 16));
}

// ---------------- f32 -> bf16 convert (vectorized x4) ----------------
__global__ __launch_bounds__(256) void cvt_bf16_kernel(const float* __restrict__ in,
                                                       bf16_t* __restrict__ out, int n4) {
  int i = blockIdx.x * 256 + threadIdx.x;
  if (i < n4) {
    v4f x = ((const v4f*)in)[i];
    v4bf y;
#pragma unroll
    for (int j = 0; j < 4; j++) y[j] = (bf16_t)x[j];
    ((v4bf*)out)[i] = y;
  }
}

// ---------------- GEMM: C[4096x1024] = A[4096x1024] @ W^T  (W row-major [1024x1024]) ----
// one wave = 32x64 C tile; ping-pong double-buffered fragments (no rotation copies)
// mode 0: bf16 out, split-head [B,H,S,64]
// mode 1: bf16 out, split-head transposed [B,H,64,S]
// mode 2: f32 out, row-major [4096,1024]
__global__ __launch_bounds__(128) void gemm_bf16_wmma(const bf16_t* __restrict__ A,
                                                      const bf16_t* __restrict__ W,
                                                      void* __restrict__ out, int mode) {
  const int K = 1024;
  int wid  = (blockIdx.x * 128 + threadIdx.x) >> 5;  // 0..2047
  int lane = threadIdx.x & 31;
  int tn = wid & 15;       // 16 n-tiles of 64
  int tm = wid >> 4;       // 128 m-tiles of 32
  int half = lane >> 4, n16 = lane & 15;

  const bf16_t* arow0 = A + (size_t)(tm * 32 + n16) * K;
  const bf16_t* arow1 = arow0 + (size_t)16 * K;
  const bf16_t* wrow  = W + (size_t)(tn * 64 + n16) * K;

  v8f c[2][4] = {};

  v16bf a0x, a1x, bx[4];   // buffer X
  v16bf a0y, a1y, by[4];   // buffer Y

  // prologue: X <- kk = 0
  a0x = lda_frag(arow0, 0, half);
  a1x = lda_frag(arow1, 0, half);
#pragma unroll
  for (int t = 0; t < 4; t++) bx[t] = ldb_frag(wrow + (size_t)t * 16 * K, 0, half);

  for (int kk = 0; kk < K; kk += 64) {
    // Y <- kk+32 (in flight during X's WMMAs)
    a0y = lda_frag(arow0, kk + 32, half);
    a1y = lda_frag(arow1, kk + 32, half);
#pragma unroll
    for (int t = 0; t < 4; t++) by[t] = ldb_frag(wrow + (size_t)t * 16 * K, kk + 32, half);

#pragma unroll
    for (int t = 0; t < 4; t++) {
      c[0][t] = wmma_bf16(a0x, bx[t], c[0][t]);
      c[1][t] = wmma_bf16(a1x, bx[t], c[1][t]);
    }

    // X <- kk+64 (wraps on last iteration; harmless reload)
    int k2 = (kk + 64) & (K - 1);
    a0x = lda_frag(arow0, k2, half);
    a1x = lda_frag(arow1, k2, half);
#pragma unroll
    for (int t = 0; t < 4; t++) bx[t] = ldb_frag(wrow + (size_t)t * 16 * K, k2, half);

#pragma unroll
    for (int t = 0; t < 4; t++) {
      c[0][t] = wmma_bf16(a0y, by[t], c[0][t]);
      c[1][t] = wmma_bf16(a1y, by[t], c[1][t]);
    }
  }

#pragma unroll
  for (int sub = 0; sub < 2; sub++) {
#pragma unroll
    for (int r = 0; r < 8; r++) {
      int mg = tm * 32 + sub * 16 + r + 8 * half;   // 0..4095
      int bidx = mg >> 10, s = mg & 1023;
#pragma unroll
      for (int t = 0; t < 4; t++) {
        int j = tn * 64 + t * 16 + n16;
        int h = j >> 6, dj = j & 63;
        float v = c[sub][t][r];
        if (mode == 0) {
          ((bf16_t*)out)[((((size_t)bidx * HH + h) * SS + s) << 6) + dj] = (bf16_t)v;
        } else if (mode == 1) {
          ((bf16_t*)out)[((((size_t)bidx * HH + h) * HD + dj) << 10) + s] = (bf16_t)v;
        } else {
          ((float*)out)[(size_t)mg * 1024 + j] = v;
        }
      }
    }
  }
}

// ---------------- flash attention with RPR ----------------
// one wave = one (head, 16-row q-tile); 2 waves per block; ping-pong pipelined key loop
__global__ __launch_bounds__(64) void attn_rpr_kernel(const bf16_t* __restrict__ qm,
                                                      const bf16_t* __restrict__ km,
                                                      const bf16_t* __restrict__ vT,
                                                      const float* __restrict__ Ek,
                                                      const float* __restrict__ Ev,
                                                      const int* __restrict__ valid_lens,
                                                      bf16_t* __restrict__ attn_out) {
  __shared__ float  ev_s[33 * 64];
  __shared__ float  qproj[2][16 * 33];
  __shared__ float  band[2][16 * 31];
  __shared__ bf16_t pbounce[2][16 * 32];
  __shared__ float  rowM[2][16], rowL[2][16], b0s[2][16], b32s[2][16];

  int tid = threadIdx.x;
  for (int i = tid; i < 33 * 64; i += 64) ev_s[i] = Ev[i];

  int w = tid >> 5, lane = tid & 31;
  int wid = blockIdx.x * 2 + w;        // 0..4095
  int bh = wid >> 6;                   // 0..63
  int qt = wid & 63;
  int b  = bh >> 4;
  int q0 = qt * 16;
  int vl = valid_lens[b];
  int half = lane >> 4, n16 = lane & 15;

  const bf16_t* qbh = qm + (size_t)bh * SS * HD;
  const bf16_t* kbh = km + (size_t)bh * SS * HD;
  const bf16_t* vbh = vT + (size_t)bh * HD * SS;

  // qproj[m][r] = q[q0+m,:] . Ek[r,:]
  for (int idx = lane; idx < 16 * 33; idx += 32) {
    int m = idx / 33, r = idx - m * 33;
    const bf16_t* qp = qbh + (size_t)(q0 + m) * HD;
    const float*  ek = Ek + r * 64;
    float acc = 0.f;
#pragma unroll 8
    for (int d = 0; d < 64; d++) acc += (float)qp[d] * ek[d];
    qproj[w][idx] = acc;
  }
  for (int idx = lane; idx < 16 * 31; idx += 32) band[w][idx] = NEGI;
  __syncthreads();

  // A-operand registers for q (K = 64 -> two K=32 chunks)
  v16bf aq0, aq1;
  {
    const bf16_t* base = qbh + (size_t)(q0 + n16) * HD;
    aq0 = cat8(ldg8(base + half * 8),      ldg8(base + half * 8 + 16));
    aq1 = cat8(ldg8(base + 32 + half * 8), ldg8(base + 32 + half * 8 + 16));
  }

  float Mx[8], L[8], B0[8];
  v8f O[4] = {v8f{}, v8f{}, v8f{}, v8f{}};
#pragma unroll
  for (int r = 0; r < 8; r++) { Mx[r] = -1.0e30f; L[r] = 0.f; B0[r] = 0.f; }

  v16bf bkA[2][2], bkB[2][2];

  auto loadK = [&](int kc, v16bf (&dst)[2][2]) {
#pragma unroll
    for (int sub = 0; sub < 2; sub++) {
      const bf16_t* kp = kbh + (size_t)(kc + sub * 16 + n16) * HD;
      dst[sub][0] = ldb_frag(kp, 0, half);
      dst[sub][1] = ldb_frag(kp, 32, half);
    }
  };

  auto chunk = [&](int kc, v16bf (&bkc)[2][2]) {
    // V fragments for this chunk, issued before the softmax work
    v16bf bvv[4];
#pragma unroll
    for (int t = 0; t < 4; t++) {
      const bf16_t* vp = vbh + (size_t)(t * 16 + n16) * SS + kc + half * 8;
      bvv[t] = cat8(ldg8(vp), ldg8(vp + 16));
    }

    float sv[2][8];
#pragma unroll
    for (int sub = 0; sub < 2; sub++) {
      int key = kc + sub * 16 + n16;
      v8f cacc = {};
      cacc = wmma_bf16(aq0, bkc[sub][0], cacc);
      cacc = wmma_bf16(aq1, bkc[sub][1], cacc);
#pragma unroll
      for (int r = 0; r < 8; r++) {
        int m = r + 8 * half;
        int qg = q0 + m;
        int d = key - qg;
        int dc = (d < -16 ? -16 : (d > 16 ? 16 : d)) + 16;
        float s = (cacc[r] + qproj[w][m * 33 + dc]) * 0.125f;
        if (key >= vl) s = NEGI;
        if (d >= -15 && d <= 15) band[w][m * 31 + (d + 15)] = s;
        sv[sub][r] = s;
      }
    }
    // online softmax update (rows: lanes 0-15 hold m=0..7, lanes 16-31 hold m=8..15)
    float f[8];
#pragma unroll
    for (int r = 0; r < 8; r++) {
      float cm = fmaxf(sv[0][r], sv[1][r]);
#pragma unroll
      for (int o = 1; o < 16; o <<= 1) cm = fmaxf(cm, __shfl_xor(cm, o, 32));
      float nM = fmaxf(Mx[r], cm);
      f[r] = __expf(Mx[r] - nM);
      Mx[r] = nM;
      float p0 = __expf(sv[0][r] - nM);
      float p1 = __expf(sv[1][r] - nM);
      int qg = q0 + r + 8 * half;
      float ps  = p0 + p1;
      float pb0 = ((kc + n16) <= qg - 16 ? p0 : 0.f) + ((kc + 16 + n16) <= qg - 16 ? p1 : 0.f);
#pragma unroll
      for (int o = 1; o < 16; o <<= 1) { ps += __shfl_xor(ps, o, 32); pb0 += __shfl_xor(pb0, o, 32); }
      L[r]  = L[r]  * f[r] + ps;
      B0[r] = B0[r] * f[r] + pb0;
      sv[0][r] = p0; sv[1][r] = p1;
#pragma unroll
      for (int t = 0; t < 4; t++) O[t][r] = O[t][r] * f[r];
    }
    // bounce P through LDS to convert C-layout -> A-layout
    __syncthreads();
#pragma unroll
    for (int sub = 0; sub < 2; sub++)
#pragma unroll
      for (int r = 0; r < 8; r++)
        pbounce[w][(r + 8 * half) * 32 + sub * 16 + n16] = (bf16_t)sv[sub][r];
    __syncthreads();
    v16bf ap;
    {
      const bf16_t* pr = &pbounce[w][n16 * 32 + half * 8];
      ap = cat8(*(const v8bf*)pr, *(const v8bf*)(pr + 16));
    }
#pragma unroll
    for (int t = 0; t < 4; t++) O[t] = wmma_bf16(ap, bvv[t], O[t]);
  };

  // ping-pong over 64-key super-chunks
  loadK(0, bkA);
  for (int kc = 0; kc < SS; kc += 64) {
    loadK(kc + 32, bkB);
    chunk(kc, bkA);
    loadK((kc + 64) & (SS - 1), bkA);
    chunk(kc + 32, bkB);
  }

  // -------- finalize: buckets + Ev contribution + store --------
  if (n16 == 0) {
#pragma unroll
    for (int r = 0; r < 8; r++) {
      int m = r + 8 * half;
      rowM[w][m] = Mx[r]; rowL[w][m] = L[r]; b0s[w][m] = B0[r] / L[r];
    }
  }
  __syncthreads();
  for (int idx = lane; idx < 16 * 31; idx += 32) {
    int m = idx / 31;
    band[w][idx] = __expf(band[w][idx] - rowM[w][m]) / rowL[w][m];
  }
  __syncthreads();
  if (lane < 16) {
    float s2 = 0.f;
#pragma unroll 8
    for (int dlt = 0; dlt < 31; dlt++) s2 += band[w][lane * 31 + dlt];
    b32s[w][lane] = 1.f - b0s[w][lane] - s2;
  }
  __syncthreads();

  int h = bh & 15;
  bf16_t* obase = attn_out + (size_t)b * SS * DM + h * HD;
#pragma unroll
  for (int r = 0; r < 8; r++) {
    int m = r + 8 * half;
    float invL = 1.f / L[r];
    float rel[4];
#pragma unroll
    for (int t = 0; t < 4; t++)
      rel[t] = b0s[w][m] * ev_s[t * 16 + n16] + b32s[w][m] * ev_s[32 * 64 + t * 16 + n16];
    for (int dlt = 0; dlt < 31; dlt++) {
      float bd = band[w][m * 31 + dlt];
#pragma unroll
      for (int t = 0; t < 4; t++) rel[t] += bd * ev_s[(dlt + 1) * 64 + t * 16 + n16];
    }
#pragma unroll
    for (int t = 0; t < 4; t++) {
      float val = O[t][r] * invL + rel[t];
      obase[(size_t)(q0 + m) * DM + t * 16 + n16] = (bf16_t)val;
    }
  }
}

extern "C" void kernel_launch(void* const* d_in, const int* in_sizes, int n_in,
                              void* d_out, int out_size, void* d_ws, size_t ws_size,
                              hipStream_t stream) {
  const float* queries    = (const float*)d_in[0];
  const float* keys       = (const float*)d_in[1];
  const float* values     = (const float*)d_in[2];
  const int*   valid_lens = (const int*)d_in[3];
  const float* Wq = (const float*)d_in[4];
  const float* Wk = (const float*)d_in[5];
  const float* Wv = (const float*)d_in[6];
  const float* Wo = (const float*)d_in[7];
  const float* Ek = (const float*)d_in[8];
  const float* Ev = (const float*)d_in[9];

  const int NX = BB * SS * DM;   // 4,194,304
  const int NW = DM * DM;        // 1,048,576

  size_t off = 0;
  auto take = [&](size_t bytes) -> void* {
    void* p = (char*)d_ws + off;
    off += (bytes + 255) & ~(size_t)255;
    return p;
  };
  bf16_t* Xq  = (bf16_t*)take((size_t)NX * 2);
  bf16_t* Xk  = (bf16_t*)take((size_t)NX * 2);
  bf16_t* Xv  = (bf16_t*)take((size_t)NX * 2);
  bf16_t* Wqb = (bf16_t*)take((size_t)NW * 2);
  bf16_t* Wkb = (bf16_t*)take((size_t)NW * 2);
  bf16_t* Wvb = (bf16_t*)take((size_t)NW * 2);
  bf16_t* Wob = (bf16_t*)take((size_t)NW * 2);
  bf16_t* Qb  = (bf16_t*)take((size_t)NX * 2);
  bf16_t* Kb  = (bf16_t*)take((size_t)NX * 2);
  bf16_t* VTb = (bf16_t*)take((size_t)NX * 2);
  bf16_t* AOb = (bf16_t*)take((size_t)NX * 2);

  cvt_bf16_kernel<<<(NX / 4 + 255) / 256, 256, 0, stream>>>(queries, Xq, NX / 4);
  cvt_bf16_kernel<<<(NX / 4 + 255) / 256, 256, 0, stream>>>(keys,    Xk, NX / 4);
  cvt_bf16_kernel<<<(NX / 4 + 255) / 256, 256, 0, stream>>>(values,  Xv, NX / 4);
  cvt_bf16_kernel<<<(NW / 4 + 255) / 256, 256, 0, stream>>>(Wq, Wqb, NW / 4);
  cvt_bf16_kernel<<<(NW / 4 + 255) / 256, 256, 0, stream>>>(Wk, Wkb, NW / 4);
  cvt_bf16_kernel<<<(NW / 4 + 255) / 256, 256, 0, stream>>>(Wv, Wvb, NW / 4);
  cvt_bf16_kernel<<<(NW / 4 + 255) / 256, 256, 0, stream>>>(Wo, Wob, NW / 4);

  // projections: 2048 waves (32x64 tiles) = 512 blocks x 128 threads
  gemm_bf16_wmma<<<512, 128, 0, stream>>>(Xq, Wqb, Qb,  0);
  gemm_bf16_wmma<<<512, 128, 0, stream>>>(Xk, Wkb, Kb,  0);
  gemm_bf16_wmma<<<512, 128, 0, stream>>>(Xv, Wvb, VTb, 1);

  // attention: 4096 q-tiles, 2 waves per block
  attn_rpr_kernel<<<2048, 64, 0, stream>>>(Qb, Kb, VTb, Ek, Ev, valid_lens, AOb);

  // output projection -> f32 d_out
  gemm_bf16_wmma<<<512, 128, 0, stream>>>(AOb, Wob, d_out, 2);
}
